// TimeTransformerEncoderLayer_44263932953360
// MI455X (gfx1250) — matrix operand order repaired and checked by reading, not compiled, not run on hardware
//
#include <hip/hip_runtime.h>
#include <hip/hip_bf16.h>

// ---------------------------------------------------------------------------
// Types for CDNA5 WMMA (wave32): bf16 16x16x32 -> f32 accumulate
// ---------------------------------------------------------------------------
typedef __bf16 bf16_t;
typedef bf16_t v8bf  __attribute__((ext_vector_type(8)));
typedef bf16_t v16bf __attribute__((ext_vector_type(16)));
typedef float  v8f   __attribute__((ext_vector_type(8)));

struct BF16x16 { v8bf lo, hi; };
static __device__ __forceinline__ v16bf join16(v8bf lo, v8bf hi) {
  BF16x16 t; t.lo = lo; t.hi = hi;
  return __builtin_bit_cast(v16bf, t);
}

static __device__ __forceinline__ float gelu_f(float x) {
  float x3 = x * x * x;
  return 0.5f * x * (1.f + tanhf(0.7978845608f * (x + 0.044715f * x3)));
}

// ---------------------------------------------------------------------------
// CDNA5 async global->LDS copy (ASYNCcnt-tracked). Guarded so the kernel
// still compiles (sync fallback) if the toolchain lacks the builtins.
// Probe result: builtin exists; param type is `int __vector(4) *` in the
// global (AS1) / LDS (AS3) address spaces.
// ---------------------------------------------------------------------------
#if defined(__has_builtin)
#if __has_builtin(__builtin_amdgcn_global_load_async_to_lds_b128)
#define USE_ASYNC_COPY 1
#endif
#endif
#ifndef USE_ASYNC_COPY
#define USE_ASYNC_COPY 0
#endif

#if USE_ASYNC_COPY
typedef int v4i_vs __attribute__((vector_size(16)));
static __device__ __forceinline__ void async_load16(const bf16_t* g, bf16_t* l) {
  // global pointer -> addrspace(1); LDS generic pointer -> 32-bit offset -> addrspace(3)
  auto gp = (__attribute__((address_space(1))) v4i_vs*)(uintptr_t)(g);
  auto lp = (__attribute__((address_space(3))) v4i_vs*)(uintptr_t)(uint32_t)(uintptr_t)(l);
  __builtin_amdgcn_global_load_async_to_lds_b128(gp, lp, 0, 0);
}
static __device__ __forceinline__ void wait_async_le(int tag) {
#if __has_builtin(__builtin_amdgcn_s_wait_asynccnt)
  if (tag) __builtin_amdgcn_s_wait_asynccnt(4);
  else     __builtin_amdgcn_s_wait_asynccnt(0);
#else
  if (tag) asm volatile("s_wait_asynccnt 0x4" ::: "memory");
  else     asm volatile("s_wait_asynccnt 0x0" ::: "memory");
#endif
}
#endif

// ---------------------------------------------------------------------------
// Generic batched bf16 WMMA GEMM:
//   C[z] = A[z] (MxK, row-major, optionally row-chunk-remapped)
//        @ Bt[z]^T (Bt is NxK row-major, i.e. pre-transposed weight)
//        (+ bias[col]) (optionally GELU), output f32 or bf16.
// Block tile: 128x128, K-step 32, double-buffered LDS, async copies.
// 256 threads = 8 waves in a 4(M) x 2(N) grid; each wave: 32x64 = 2x4 WMMAs.
// Assumes: M % 128 == 0, K % 32 == 0 (true for every call in this layer).
// Partial N handled by clamping B rows (garbage feeds only unstored columns).
// A row remap:  addr = Abase + (r / aRPC) * aCS + (r % aRPC) * lda
// C row remap:  off  = cOffZ + (r / cRPC) * cCS + (r % cRPC) * ldc + col
// C z offset:   cOffZ = (z / zDivC) * zCHi + (z % zDivC) * zCLo
// ---------------------------------------------------------------------------
#define GBK 32
#define SAK 40   // padded LDS row stride (elements): 80B rows, 16B aligned

__global__ __launch_bounds__(256) void gemm_bf16_wmma_kernel(
    const bf16_t* __restrict__ A, const bf16_t* __restrict__ Bt,
    const float* __restrict__ bias, void* __restrict__ C,
    int M, int N, int K,
    long zStrideA, long zStrideB,
    int zDivC, long zCHi, long zCLo,
    int lda, int ldb, int ldc,
    int aRPC, long aCS,
    int cRPC, long cCS,
    long zStrideBias,
    int epilogue,      // 0 = none, 1 = +bias, 2 = +bias then gelu
    int outBf16)       // 0 = f32 C, 1 = bf16 C
{
  __shared__ __align__(16) bf16_t sA[2][128 * SAK];
  __shared__ __align__(16) bf16_t sB[2][128 * SAK];

  const int z = blockIdx.z;
  const bf16_t* Ab = A + (long)z * zStrideA;
  const bf16_t* Bb = Bt + (long)z * zStrideB;
  const long cOffZ = (long)(z / zDivC) * zCHi + (long)(z % zDivC) * zCLo;
  const float* biasb = bias ? (bias + (long)z * zStrideBias) : nullptr;

  const int tid  = threadIdx.x;
  const int lane = tid & 31;
  const int wave = tid >> 5;
  const int wm = wave >> 1;          // 0..3
  const int wn = wave & 1;           // 0..1
  const int m0 = blockIdx.y * 128;
  const int n0 = blockIdx.x * 128;
  const int lhalf = lane >> 4;       // 0 or 1
  const int l15   = lane & 15;

  // Per-thread staging coordinates: 512 16B chunks per tile, 2 per thread.
  // chunk c: row r = c>>2, col offset kc = (c&3)*8 elements.
  const int ar0 = tid >> 2,          ak0 = (tid & 3) * 8;
  const int ar1 = (tid + 256) >> 2,  ak1 = (tid & 3) * 8;  // same kc pattern

  // Resolve A row addresses once (row-chunk remap outside the hot loop).
  const int arow0 = m0 + ar0, arow1 = m0 + ar1;
  const bf16_t* aRow0 = Ab + (long)(arow0 / aRPC) * aCS + (long)(arow0 % aRPC) * lda;
  const bf16_t* aRow1 = Ab + (long)(arow1 / aRPC) * aCS + (long)(arow1 % aRPC) * lda;
  // B rows clamped into range (partial-N tiles).
  const int bcol0 = (n0 + ar0 < N) ? (n0 + ar0) : (N - 1);
  const int bcol1 = (n0 + ar1 < N) ? (n0 + ar1) : (N - 1);
  const bf16_t* bRow0 = Bb + (long)bcol0 * ldb;
  const bf16_t* bRow1 = Bb + (long)bcol1 * ldb;

  auto stage = [&](int k0, int buf) {
#if USE_ASYNC_COPY
    async_load16(aRow0 + k0 + ak0, &sA[buf][ar0 * SAK + ak0]);
    async_load16(aRow1 + k0 + ak1, &sA[buf][ar1 * SAK + ak1]);
    async_load16(bRow0 + k0 + ak0, &sB[buf][ar0 * SAK + ak0]);
    async_load16(bRow1 + k0 + ak1, &sB[buf][ar1 * SAK + ak1]);
#else
    *(v8bf*)&sA[buf][ar0 * SAK + ak0] = *(const v8bf*)(aRow0 + k0 + ak0);
    *(v8bf*)&sA[buf][ar1 * SAK + ak1] = *(const v8bf*)(aRow1 + k0 + ak1);
    *(v8bf*)&sB[buf][ar0 * SAK + ak0] = *(const v8bf*)(bRow0 + k0 + ak0);
    *(v8bf*)&sB[buf][ar1 * SAK + ak1] = *(const v8bf*)(bRow1 + k0 + ak1);
#endif
    __builtin_prefetch(aRow0 + k0 + GBK + ak0, 0, 1);   // global_prefetch_b8
    __builtin_prefetch(bRow0 + k0 + GBK + ak0, 0, 1);
  };

  v8f acc[2][4];
  for (int i = 0; i < 2; ++i)
    for (int j = 0; j < 4; ++j)
      for (int r = 0; r < 8; ++r) acc[i][j][r] = 0.f;

  const int ntiles = K / GBK;
  stage(0, 0);

  for (int it = 0; it < ntiles; ++it) {
    const int cur = it & 1;
    const int more = (it + 1 < ntiles);
    if (more) stage((it + 1) * GBK, cur ^ 1);   // overlap next tile with compute
#if USE_ASYNC_COPY
    wait_async_le(more);                        // tile `it` fully in LDS (this wave)
#endif
    __syncthreads();                            // ... and for all waves

    // ---- fragments per ISA VGPR layouts ----
    // A (16x32 bf16): lane l, elem e -> m = l&15, k = 16*(e>>3) + 8*(l>>4) + (e&7)
    // B (32x16 bf16, from Bt): lane l, elem e -> k = 16*(e>>3) + 8*(l>>4) + (e&7), n = l&15
    v16bf afrag[2], bfrag[4];
    const int kc8 = lhalf * 8;
    const int ks  = lhalf * 16;
    for (int tm = 0; tm < 2; ++tm) {
      int m = wm * 32 + tm * 16 + l15;
      v8bf lo = *(const v8bf*)&sA[cur][m * SAK + kc8];
      v8bf hi = *(const v8bf*)&sA[cur][m * SAK + 16 + kc8];
      afrag[tm] = join16(lo, hi);
    }
    for (int tn = 0; tn < 4; ++tn) {
      int n = wn * 64 + tn * 16 + l15;
      v8bf lo = *(const v8bf*)&sB[cur][n * SAK + kc8];
      v8bf hi = *(const v8bf*)&sB[cur][n * SAK + 16 + kc8];
      bfrag[tn] = join16(lo, hi);
    }

    for (int tm = 0; tm < 2; ++tm)
      for (int tn = 0; tn < 4; ++tn)
        acc[tm][tn] = __builtin_amdgcn_wmma_f32_16x16x32_bf16(
            false, afrag[tm], false, bfrag[tn],
            (short)0, acc[tm][tn], false, false);

    __syncthreads();                            // all waves done reading buf `cur`
  }

  // ---- epilogue + store ----
  int cols[4]; float bvv[4];
  for (int tn = 0; tn < 4; ++tn) {
    cols[tn] = n0 + wn * 64 + tn * 16 + l15;
    bvv[tn] = (epilogue != 0 && biasb && cols[tn] < N) ? biasb[cols[tn]] : 0.f;
  }
  for (int tm = 0; tm < 2; ++tm) {
    for (int r = 0; r < 8; ++r) {
      int row = m0 + wm * 32 + tm * 16 + lhalf * 8 + r;
      long rowOff = cOffZ + (long)(row / cRPC) * cCS + (long)(row % cRPC) * ldc;
      for (int tn = 0; tn < 4; ++tn) {
        if (cols[tn] >= N) continue;
        float v = acc[tm][tn][r] + bvv[tn];
        if (epilogue == 2) v = gelu_f(v);
        long off = rowOff + cols[tn];
        if (outBf16) ((bf16_t*)C)[off] = (bf16_t)v;
        else         ((float*)C)[off]  = v;
      }
    }
  }
}

// ---------------------------------------------------------------------------
// Elementwise f32 -> bf16
// ---------------------------------------------------------------------------
__global__ void cvt_f32_bf16_kernel(const float* __restrict__ in,
                                    bf16_t* __restrict__ out, long n) {
  long i = (long)blockIdx.x * blockDim.x + threadIdx.x;
  if (i < n) out[i] = (bf16_t)in[i];
}

// ---------------------------------------------------------------------------
// Tiled transpose with conversion to bf16: in [z][R][C] -> out [z][C][R]
// ---------------------------------------------------------------------------
template <typename T>
__global__ void transpose_to_bf16_kernel(const T* __restrict__ in,
                                         bf16_t* __restrict__ out,
                                         int R, int C, long inZ, long outZ) {
  __shared__ bf16_t tile[32][33];
  const T* ib = in + (long)blockIdx.z * inZ;
  bf16_t*  ob = out + (long)blockIdx.z * outZ;
  int c0 = blockIdx.x * 32, r0 = blockIdx.y * 32;
  int tx = threadIdx.x & 31, ty = threadIdx.x >> 5;   // 256 threads
  for (int i = 0; i < 4; ++i) {
    int r = r0 + ty + i * 8, c = c0 + tx;
    if (r < R && c < C) tile[ty + i * 8][tx] = (bf16_t)(float)ib[(long)r * C + c];
  }
  __syncthreads();
  for (int i = 0; i < 4; ++i) {
    int orow = c0 + ty + i * 8;   // output row = original column
    int ocol = r0 + tx;           // output col = original row
    if (orow < C && ocol < R) ob[(long)orow * R + ocol] = tile[tx][ty + i * 8];
  }
}

// ---------------------------------------------------------------------------
// Reshape q/kv GEMM outputs into per-head layouts:
//   qbf [b][h][n][d], kbf [b][h][n][d], vtbf [b][h][d][n]   (b=4,h=16,n=1024,d=64)
// ---------------------------------------------------------------------------
__global__ void qkv_reshape_kernel(const bf16_t* __restrict__ qbuf,
                                   const bf16_t* __restrict__ kvbuf,
                                   bf16_t* __restrict__ qbf,
                                   bf16_t* __restrict__ kbf,
                                   bf16_t* __restrict__ vtbf) {
  long i = (long)blockIdx.x * blockDim.x + threadIdx.x;  // [b][h][n][d]
  int d = (int)(i & 63);  long t = i >> 6;
  int n = (int)(t & 1023); t >>= 10;
  int h = (int)(t & 15);
  int b = (int)(t >> 4);
  long rowBN = (long)(b * 1024 + n);
  qbf[i] = qbuf[rowBN * 1024 + h * 64 + d];
  kbf[i] = kvbuf[rowBN * 2048 + h * 64 + d];
  vtbf[((long)((b * 16 + h) * 64 + d)) * 1024 + n] = kvbuf[rowBN * 2048 + 1024 + h * 64 + d];
}

// ---------------------------------------------------------------------------
// Row softmax over `cols` (<=2048), scaled input, bf16 output. One block/row.
// ---------------------------------------------------------------------------
__global__ void row_softmax_kernel(const float* __restrict__ in,
                                   bf16_t* __restrict__ out,
                                   int cols, float scale) {
  long row = blockIdx.x;
  const float* p = in + row * (long)cols;
  float vals[8];
  int nv = 0;
  float lmax = -3.4e38f;
  for (int c = threadIdx.x; c < cols; c += 256) {
    float v = p[c] * scale;
    vals[nv++] = v;
    lmax = fmaxf(lmax, v);
  }
  __shared__ float red[256];
  red[threadIdx.x] = lmax; __syncthreads();
  for (int s = 128; s > 0; s >>= 1) {
    if (threadIdx.x < s) red[threadIdx.x] = fmaxf(red[threadIdx.x], red[threadIdx.x + s]);
    __syncthreads();
  }
  float m = red[0]; __syncthreads();
  float lsum = 0.f;
  for (int i = 0; i < nv; ++i) { vals[i] = __expf(vals[i] - m); lsum += vals[i]; }
  red[threadIdx.x] = lsum; __syncthreads();
  for (int s = 128; s > 0; s >>= 1) {
    if (threadIdx.x < s) red[threadIdx.x] += red[threadIdx.x + s];
    __syncthreads();
  }
  float inv = 1.f / red[0];
  nv = 0;
  for (int c = threadIdx.x; c < cols; c += 256)
    out[row * (long)cols + c] = (bf16_t)(vals[nv++] * inv);
}

// ---------------------------------------------------------------------------
// Column softmax (dispatch): logits [b][n][es] -> softmax over n,
// written transposed as dispT [b][es][n] in bf16. One thread per column.
// ---------------------------------------------------------------------------
__global__ void col_softmax_T_kernel(const float* __restrict__ logits,
                                     bf16_t* __restrict__ dispT,
                                     int nrows, int ncols) {
  int b = blockIdx.y;
  int col = blockIdx.x * 256 + threadIdx.x;
  if (col >= ncols) return;
  const float* base = logits + (long)b * nrows * ncols;
  float m = -3.4e38f;
  for (int n = 0; n < nrows; ++n) m = fmaxf(m, base[(long)n * ncols + col]);
  float s = 0.f;
  for (int n = 0; n < nrows; ++n) s += __expf(base[(long)n * ncols + col] - m);
  float inv = 1.f / s;
  bf16_t* o = dispT + (long)b * ncols * nrows + (long)col * nrows;
  for (int n = 0; n < nrows; ++n)
    o[n] = (bf16_t)(__expf(base[(long)n * ncols + col] - m) * inv);
}

// ---------------------------------------------------------------------------
// Attention-probability transpose (the layer's largest memory op, 256MB):
//   pbf [b][h][n][m] (bf16) -> attn f32 [b][n][m][h]
// LDS-tiled so every lane emits 64B of contiguous float4 stores.
// Grid: (SEQ/256, SEQ, B), 256 threads.
// ---------------------------------------------------------------------------
__global__ void attn_transpose_kernel(const bf16_t* __restrict__ pbf,
                                      float* __restrict__ attn) {
  __shared__ float tile[16][257];
  const int m0 = blockIdx.x * 256;
  const int n  = blockIdx.y;
  const int b  = blockIdx.z;
  const int t  = threadIdx.x;
  for (int h = 0; h < 16; ++h)
    tile[h][t] = (float)pbf[(((long)(b * 16 + h) * 1024 + n) * 1024) + m0 + t];
  __syncthreads();
  float4* out4 = (float4*)(attn + (((long)b * 1024 + n) * 1024 + m0) * 16);
  for (int j = 0; j < 4; ++j) {
    float4 v;
    v.x = tile[j * 4 + 0][t];
    v.y = tile[j * 4 + 1][t];
    v.z = tile[j * 4 + 2][t];
    v.w = tile[j * 4 + 3][t];
    out4[t * 4 + j] = v;
  }
}

// ---------------------------------------------------------------------------
// Residual + LayerNorm:  v = a + beta*b ;  out = (v-mu)*rsqrt(var+eps)*g + bb
// Optionally also writes bf16 copy. One block per row (cols <= 2048).
// ---------------------------------------------------------------------------
__global__ void ln_residual_kernel(const float* __restrict__ a,
                                   const float* __restrict__ b, float beta,
                                   const float* __restrict__ g,
                                   const float* __restrict__ bb,
                                   float* __restrict__ outF,
                                   bf16_t* __restrict__ outB, int cols) {
  long row = blockIdx.x;
  const float* pa = a + row * (long)cols;
  const float* pb = b + row * (long)cols;
  float vals[8]; int nv = 0;
  float s = 0.f, s2 = 0.f;
  for (int c = threadIdx.x; c < cols; c += 256) {
    float v = pa[c] + beta * pb[c];
    vals[nv++] = v; s += v; s2 += v * v;
  }
  __shared__ float red[256];
  red[threadIdx.x] = s; __syncthreads();
  for (int t = 128; t > 0; t >>= 1) {
    if (threadIdx.x < t) red[threadIdx.x] += red[threadIdx.x + t];
    __syncthreads();
  }
  float mu = red[0] / cols; __syncthreads();
  red[threadIdx.x] = s2; __syncthreads();
  for (int t = 128; t > 0; t >>= 1) {
    if (threadIdx.x < t) red[threadIdx.x] += red[threadIdx.x + t];
    __syncthreads();
  }
  float var = red[0] / cols - mu * mu;
  float rstd = rsqrtf(var + 1e-5f);
  nv = 0;
  for (int c = threadIdx.x; c < cols; c += 256) {
    float y = (vals[nv++] - mu) * rstd * g[c] + bb[c];
    if (outF) outF[row * (long)cols + c] = y;
    if (outB) outB[row * (long)cols + c] = (bf16_t)y;
  }
}

// ---------------------------------------------------------------------------
// Host orchestration
// ---------------------------------------------------------------------------
extern "C" void kernel_launch(void* const* d_in, const int* in_sizes, int n_in,
                              void* d_out, int out_size, void* d_ws, size_t ws_size,
                              hipStream_t stream) {
  (void)in_sizes; (void)n_in; (void)out_size; (void)ws_size;

  constexpr int B = 4, SEQ = 1024, DIM = 1024, H = 16, HD = 64;
  constexpr int HID = 4096, EXP = 2, SLOTS = 512, ES = EXP * SLOTS;     // 1024
  constexpr long M  = (long)B * SEQ;                                    // 4096
  constexpr int BIG = 1 << 30;

  const float* x    = (const float*)d_in[0];
  const float* Wq   = (const float*)d_in[1];
  const float* bq   = (const float*)d_in[2];
  const float* Wkv  = (const float*)d_in[3];
  const float* bkv  = (const float*)d_in[4];
  const float* Wp   = (const float*)d_in[5];
  const float* bp   = (const float*)d_in[6];
  const float* g1   = (const float*)d_in[7];
  const float* b1   = (const float*)d_in[8];
  const float* phi  = (const float*)d_in[9];
  const float* We1  = (const float*)d_in[10];
  const float* be1  = (const float*)d_in[11];
  const float* We2  = (const float*)d_in[12];
  const float* be2  = (const float*)d_in[13];
  const float* g2   = (const float*)d_in[14];
  const float* b2   = (const float*)d_in[15];

  float* y_out    = (float*)d_out;                       // [B][SEQ][DIM]
  float* attn_out = (float*)d_out + M * DIM;             // [B][SEQ][SEQ][H]
  float* scores   = attn_out;                            // staging: [b*h][n][m] f32

  // Workspace bump allocator (256B aligned)
  char* wsp = (char*)d_ws;
  size_t off = 0;
  auto alloc = [&](size_t bytes) -> void* {
    off = (off + 255) & ~(size_t)255;
    void* p = wsp + off;
    off += bytes;
    return p;
  };
  bf16_t* xbf     = (bf16_t*)alloc(M * DIM * 2);
  bf16_t* wqt     = (bf16_t*)alloc((size_t)DIM * DIM * 2);
  bf16_t* wkvt    = (bf16_t*)alloc((size_t)2 * DIM * DIM * 2);
  bf16_t* wpt     = (bf16_t*)alloc((size_t)DIM * DIM * 2);
  bf16_t* phit    = (bf16_t*)alloc((size_t)DIM * ES * 2);
  bf16_t* we1t    = (bf16_t*)alloc((size_t)EXP * HID * DIM * 2);
  bf16_t* we2t    = (bf16_t*)alloc((size_t)EXP * DIM * HID * 2);
  bf16_t* qbuf    = (bf16_t*)alloc(M * DIM * 2);
  bf16_t* kvbuf   = (bf16_t*)alloc(M * 2 * DIM * 2);
  bf16_t* qbf     = (bf16_t*)alloc(M * DIM * 2);
  bf16_t* kbf     = (bf16_t*)alloc(M * DIM * 2);
  bf16_t* vtbf    = (bf16_t*)alloc(M * DIM * 2);
  bf16_t* pbf     = (bf16_t*)alloc((size_t)B * H * SEQ * SEQ * 2);     // 128MB
  bf16_t* attnob  = (bf16_t*)alloc(M * DIM * 2);
  float*  projb   = (float*)alloc(M * DIM * 4);
  float*  x1f     = (float*)alloc(M * DIM * 4);
  bf16_t* x1bf    = (bf16_t*)alloc(M * DIM * 2);
  bf16_t* x1t     = (bf16_t*)alloc(M * DIM * 2);                       // [b][d][n]
  float*  logitsf = (float*)alloc(M * ES * 4);
  bf16_t* combbf  = (bf16_t*)alloc(M * ES * 2);
  bf16_t* dispT   = (bf16_t*)alloc((size_t)B * ES * SEQ * 2);          // [b][es][n]
  bf16_t* slotsbf = (bf16_t*)alloc((size_t)B * ES * DIM * 2);          // [b][es][d]
  bf16_t* hbf     = (bf16_t*)alloc((size_t)EXP * B * SLOTS * HID * 2); // [e][b*s][hid]
  bf16_t* yexpbf  = (bf16_t*)alloc((size_t)B * ES * DIM * 2);          // [b][es][d]
  bf16_t* yexpT   = (bf16_t*)alloc((size_t)B * DIM * ES * 2);          // [b][d][es]
  float*  moebuf  = (float*)alloc(M * DIM * 4);

  auto gemm = [&](const bf16_t* A, const bf16_t* Bt, const float* bias, void* C,
                  int Mm, int Nn, int Kk, int Z,
                  long zA, long zB, int zDivC, long zCHi, long zCLo,
                  int lda, int ldb, int ldc,
                  int aRPC, long aCS, int cRPC, long cCS, long zBias,
                  int epi, int obf) {
    dim3 grid((Nn + 127) / 128, (Mm + 127) / 128, Z);
    gemm_bf16_wmma_kernel<<<grid, dim3(256), 0, stream>>>(
        A, Bt, bias, C, Mm, Nn, Kk, zA, zB, zDivC, zCHi, zCLo,
        lda, ldb, ldc, aRPC, aCS, cRPC, cCS, zBias, epi, obf);
  };

  // 1) x -> bf16
  cvt_f32_bf16_kernel<<<(int)(M * DIM / 256), 256, 0, stream>>>(x, xbf, M * DIM);

  // 2) weight transposes (f32 -> bf16, [K][N] -> [N][K])
  transpose_to_bf16_kernel<float><<<dim3(32, 32, 1), 256, 0, stream>>>(Wq,  wqt,  DIM, DIM, 0, 0);
  transpose_to_bf16_kernel<float><<<dim3(64, 32, 1), 256, 0, stream>>>(Wkv, wkvt, DIM, 2 * DIM, 0, 0);
  transpose_to_bf16_kernel<float><<<dim3(32, 32, 1), 256, 0, stream>>>(Wp,  wpt,  DIM, DIM, 0, 0);
  transpose_to_bf16_kernel<float><<<dim3(32, 32, 1), 256, 0, stream>>>(phi, phit, DIM, ES, 0, 0);
  transpose_to_bf16_kernel<float><<<dim3(128, 32, 2), 256, 0, stream>>>(We1, we1t, DIM, HID,
      (long)DIM * HID, (long)HID * DIM);
  transpose_to_bf16_kernel<float><<<dim3(32, 128, 2), 256, 0, stream>>>(We2, we2t, HID, DIM,
      (long)HID * DIM, (long)DIM * HID);

  // 3) q = x@Wq + bq   ;  kv = x@Wkv + bkv      (bf16 out)
  gemm(xbf, wqt,  bq,  qbuf,  (int)M, DIM,     DIM, 1, 0, 0, 1, 0, 0,
       DIM, DIM, DIM, BIG, 0, BIG, 0, 0, 1, 1);
  gemm(xbf, wkvt, bkv, kvbuf, (int)M, 2 * DIM, DIM, 1, 0, 0, 1, 0, 0,
       DIM, DIM, 2 * DIM, BIG, 0, BIG, 0, 0, 1, 1);

  // 4) per-head layouts
  qkv_reshape_kernel<<<(int)(M * DIM / 256), 256, 0, stream>>>(qbuf, kvbuf, qbf, kbf, vtbf);

  // 5) scores[z=b*h] = Q_h @ K_h^T  -> staged in d_out attn region (f32)
  gemm(qbf, kbf, nullptr, scores, SEQ, SEQ, HD, B * H,
       (long)SEQ * HD, (long)SEQ * HD, 1, (long)SEQ * SEQ, 0,
       HD, HD, SEQ, BIG, 0, BIG, 0, 0, 0, 0);

  // 6) softmax over m (scale = 1/sqrt(hd)) -> pbf bf16 [b][h][n][m]
  row_softmax_kernel<<<B * H * SEQ, 256, 0, stream>>>(scores, pbf, SEQ, 0.125f);

  // 7) transpose probabilities to output layout [b][n][m][h] (f32, coalesced)
  attn_transpose_kernel<<<dim3(SEQ / 256, SEQ, B), 256, 0, stream>>>(pbf, attn_out);

  // 8) attn@V: per (b,h)  P(1024x1024) @ Vt(64x1024)^T -> attnob [b][n][h*64+d]
  gemm(pbf, vtbf, nullptr, attnob, SEQ, HD, SEQ, B * H,
       (long)SEQ * SEQ, (long)HD * SEQ, H, (long)SEQ * DIM, HD,
       SEQ, SEQ, DIM, BIG, 0, BIG, 0, 0, 0, 1);

  // 9) out-proj: attnob @ Wp + bp (f32)
  gemm(attnob, wpt, bp, projb, (int)M, DIM, DIM, 1, 0, 0, 1, 0, 0,
       DIM, DIM, DIM, BIG, 0, BIG, 0, 0, 1, 0);

  // 10) x1 = LN(proj + x)
  ln_residual_kernel<<<(int)M, 256, 0, stream>>>(projb, x, 1.0f, g1, b1, x1f, x1bf, DIM);

  // 11) logits = x1 @ phi (f32)
  gemm(x1bf, phit, nullptr, logitsf, (int)M, ES, DIM, 1, 0, 0, 1, 0, 0,
       DIM, DIM, ES, BIG, 0, BIG, 0, 0, 0, 0);

  // 12) combine = row-softmax(logits) (bf16) ; dispatch = col-softmax, transposed
  row_softmax_kernel<<<(int)M, 256, 0, stream>>>(logitsf, combbf, ES, 1.0f);
  col_softmax_T_kernel<<<dim3(ES / 256, B, 1), 256, 0, stream>>>(logitsf, dispT, SEQ, ES);

  // 13) x1^T per batch: [b][d][n]
  transpose_to_bf16_kernel<float><<<dim3(32, 32, B), 256, 0, stream>>>(
      x1f, x1t, SEQ, DIM, (long)SEQ * DIM, (long)DIM * SEQ);

  // 14) slots[b] = dispatch^T @ x1 : [es][n] @ [n][d] (Bt = x1t) -> bf16
  gemm(dispT, x1t, nullptr, slotsbf, ES, DIM, SEQ, B,
       (long)ES * SEQ, (long)DIM * SEQ, 1, (long)ES * DIM, 0,
       SEQ, SEQ, DIM, BIG, 0, BIG, 0, 0, 0, 1);

  // 15) h[e] = gelu(slots_e @ We1_e + be1_e): A rows gathered from [b][es][d]
  gemm(slotsbf, we1t, be1, hbf, B * SLOTS, HID, DIM, EXP,
       (long)SLOTS * DIM, (long)HID * DIM, 1, (long)B * SLOTS * HID, 0,
       DIM, DIM, HID, SLOTS, (long)ES * DIM, BIG, 0, HID, 2, 1);

  // 16) yexp = h @ We2 + be2, scattered back to [b][es][d]
  gemm(hbf, we2t, be2, yexpbf, B * SLOTS, DIM, HID, EXP,
       (long)B * SLOTS * HID, (long)DIM * HID, 1, (long)SLOTS * DIM, 0,
       HID, HID, DIM, BIG, 0, SLOTS, (long)ES * DIM, DIM, 1, 1);

  // 17) yexp^T per batch: [b][d][es]
  transpose_to_bf16_kernel<bf16_t><<<dim3(32, 32, B), 256, 0, stream>>>(
      yexpbf, yexpT, ES, DIM, (long)ES * DIM, (long)DIM * ES);

  // 18) moe = combine @ yexp : [n][es] @ (Bt=[d][es]) per batch (f32)
  gemm(combbf, yexpT, nullptr, moebuf, SEQ, DIM, ES, B,
       (long)SEQ * ES, (long)DIM * ES, 1, (long)SEQ * DIM, 0,
       ES, ES, DIM, BIG, 0, BIG, 0, 0, 0, 0);

  // 19) y = LN(moe + 2*x1)  -> d_out
  ln_residual_kernel<<<(int)M, 256, 0, stream>>>(moebuf, x1f, 2.0f, g2, b2, y_out, nullptr, DIM);
}
